// LightDropoutLSTM_66563403153605
// MI455X (gfx1250) — matrix operand order repaired and checked by reading, not compiled, not run on hardware
//
#include <hip/hip_runtime.h>
#include <stdint.h>

// ---------------------------------------------------------------------------
// Fused 3-layer LSTM (B=1024, T=512, D=128 -> 64 -> 32 -> 16) + FC(16->1).
// One workgroup (128 threads = 4 wave32) owns a 16-row batch tile and walks
// all 512 timesteps with everything (weights, h, c, x_t) resident in LDS /
// registers. Matmuls use v_wmma_f32_16x16x32_f16 (f16 in, f32 accumulate).
// Only HBM traffic: x read once (256 MB) + 1024 output floats.
// ---------------------------------------------------------------------------

typedef __attribute__((ext_vector_type(16))) _Float16 v16h;
typedef __attribute__((ext_vector_type(8)))  float    v8f;

#define TPB     128
#define B_TILE  16
#define T_STEPS 512
#define D_IN    128

union V16 { v16h v; uint4 q[2]; };

__device__ __forceinline__ v8f wmma_f16(v16h a, v16h b, v8f c) {
  // D = A(16x32,f16) * B(32x16,f16) + C(16x16,f32)
  return __builtin_amdgcn_wmma_f32_16x16x32_f16(
      /*neg_a=*/false, a, /*neg_b=*/false, b,
      /*c_mod=*/(short)0, c, /*reuse_a=*/false, /*reuse_b=*/false);
}

// A-fragment (16 rows x 32 K) from a row-major f16 LDS buffer.
// ISA 16-bit A layout: lane l -> row = l&15; lanes 0-15 hold K = kt*32+[0..7]
// in halfs 0..7 and K = kt*32+[16..23] in halfs 8..15; lanes 16-31 the same
// shifted by 8.  Both chunks are 16 contiguous bytes -> two ds_load_b128.
__device__ __forceinline__ v16h ldsA(const _Float16* buf, int strideH, int kt, int lane) {
  const _Float16* p = buf + (lane & 15) * strideH + kt * 32 + ((lane >> 4) << 3);
  V16 u;
  u.q[0] = *reinterpret_cast<const uint4*>(p);
  u.q[1] = *reinterpret_cast<const uint4*>(p + 16);
  return u.v;
}

// B-fragment from pre-swizzled LDS weights: frag-major, then lane-major,
// 16 contiguous halfs (32 bytes) per lane.
__device__ __forceinline__ v16h ldsB(const _Float16* wf, int frag, int lane) {
  const _Float16* p = wf + ((frag * 32 + lane) << 4);
  V16 u;
  u.q[0] = *reinterpret_cast<const uint4*>(p);
  u.q[1] = *reinterpret_cast<const uint4*>(p + 8);
  return u.v;
}

__device__ __forceinline__ float fsigmoid(float x) {
  return __builtin_amdgcn_rcpf(1.0f + __expf(-x));
}
__device__ __forceinline__ float ftanh(float x) {
  float e = __expf(-2.0f * x);
  return (1.0f - e) * __builtin_amdgcn_rcpf(1.0f + e);
}

// Transpose + f16-quantize + B-fragment-swizzle a torch-layout weight
// W[4H][KIN] (gates = X @ W^T, so B[k][n] = W[n][k]) into LDS.
// B-fragment layout: lane l -> col n = nt*16 + (l&15); halfs h hold
// K rows kt*32 + 16*(h>>3) + 8*(l>>4) + (h&7).  Rows k >= KIN are zero padding.
template <int KT, int NT, int KIN>
__device__ __forceinline__ void swizzleW(_Float16* dst, const float* W, int tid) {
  const int total = KT * NT * 512;
  for (int idx = tid; idx < total; idx += TPB) {
    int h    = idx & 15;
    int ln   = (idx >> 4) & 31;
    int frag = idx >> 9;
    int kt   = frag / NT;
    int nt   = frag - kt * NT;
    int k    = kt * 32 + ((h >> 3) << 4) + ((ln >> 4) << 3) + (h & 7);
    int n    = nt * 16 + (ln & 15);
    float v  = (k < KIN) ? W[n * KIN + k] : 0.0f;
    dst[idx] = (_Float16)v;
  }
}

__global__ __launch_bounds__(TPB, 1)
void lstm3_fused_kernel(const float* __restrict__ x,
                        const float* __restrict__ Wih1, const float* __restrict__ Whh1,
                        const float* __restrict__ bih1, const float* __restrict__ bhh1,
                        const float* __restrict__ Wih2, const float* __restrict__ Whh2,
                        const float* __restrict__ bih2, const float* __restrict__ bhh2,
                        const float* __restrict__ Wih3, const float* __restrict__ Whh3,
                        const float* __restrict__ bih3, const float* __restrict__ bhh3,
                        const float* __restrict__ Wfc,  const float* __restrict__ bfc,
                        float* __restrict__ out) {
  // ---- LDS: swizzled f16 weights (~128 KB) + activations (~10 KB) ----
  __shared__ __align__(16) _Float16 Wf_ih1[64 * 512];  // K=128 (KT=4), N=256 (NT=16)
  __shared__ __align__(16) _Float16 Wf_hh1[32 * 512];  // K=64  (KT=2), N=256
  __shared__ __align__(16) _Float16 Wf_ih2[16 * 512];  // K=64  (KT=2), N=128 (NT=8)
  __shared__ __align__(16) _Float16 Wf_hh2[ 8 * 512];  // K=32  (KT=1), N=128
  __shared__ __align__(16) _Float16 Wf_ih3[ 4 * 512];  // K=32  (KT=1), N=64  (NT=4)
  __shared__ __align__(16) _Float16 Wf_hh3[ 4 * 512];  // K=16->pad32,  N=64
  __shared__ float bias1[256];
  __shared__ float bias2[128];
  __shared__ float bias3[64];
  __shared__ __align__(16) _Float16 xbuf [16 * 128];
  __shared__ __align__(16) _Float16 h1buf[16 * 64];
  __shared__ __align__(16) _Float16 h2buf[16 * 32];
  __shared__ __align__(16) _Float16 h3buf[16 * 32];   // cols 16..31 stay zero (K pad)

  const int tid  = threadIdx.x;
  const int lane = tid & 31;
  const int wv   = tid >> 5;          // wave id 0..3
  const int b0   = blockIdx.x * B_TILE;

  // ---- one-time setup: quantize+swizzle weights, fold biases, zero states ----
  swizzleW<4, 16, 128>(Wf_ih1, Wih1, tid);
  swizzleW<2, 16,  64>(Wf_hh1, Whh1, tid);
  swizzleW<2,  8,  64>(Wf_ih2, Wih2, tid);
  swizzleW<1,  8,  32>(Wf_hh2, Whh2, tid);
  swizzleW<1,  4,  32>(Wf_ih3, Wih3, tid);
  swizzleW<1,  4,  16>(Wf_hh3, Whh3, tid);
  for (int i = tid; i < 256; i += TPB) bias1[i] = bih1[i] + bhh1[i];
  for (int i = tid; i < 128; i += TPB) bias2[i] = bih2[i] + bhh2[i];
  for (int i = tid; i <  64; i += TPB) bias3[i] = bih3[i] + bhh3[i];
  for (int i = tid; i < 16 * 64; i += TPB) h1buf[i] = (_Float16)0.0f;
  for (int i = tid; i < 16 * 32; i += TPB) h2buf[i] = (_Float16)0.0f;
  for (int i = tid; i < 16 * 32; i += TPB) h3buf[i] = (_Float16)0.0f;
  __syncthreads();

  // ---- layer-1 recurrent-weight B fragments held in registers for the whole
  // T loop (critical path, used by all 4 waves every step).  Layer-2/3
  // recurrent fragments are streamed from LDS inside their phases to keep the
  // wave under 256 VGPRs (avoids s_set_vgpr_msb churn in the inner loop).
  v16h bh1[2][4];
#pragma unroll
  for (int kt = 0; kt < 2; ++kt)
#pragma unroll
    for (int G = 0; G < 4; ++G)
      bh1[kt][G] = ldsB(Wf_hh1, kt * 16 + G * 4 + wv, lane);

  // cell states: v8f-per-lane matches WMMA D layout (row = r + 8*(lane>=16))
  float c1[8], c2[8], c3[8];
#pragma unroll
  for (int r = 0; r < 8; ++r) { c1[r] = 0.f; c2[r] = 0.f; c3[r] = 0.f; }

  // x staging plan: thread (tid) loads row m = tid>>3, 16 floats at col (tid&7)*16
  const int xm = tid >> 3;
  const int xk = (tid & 7) << 4;
  const float* xsrc = x + (size_t)(b0 + xm) * (T_STEPS * D_IN) + xk;
  _Float16*    xdst = xbuf + xm * D_IN + xk;
  const int rbase = (lane >> 4) << 3;

  for (int t = 0; t < T_STEPS; ++t) {
    // ---- phase 0: stage x_t (fp32->f16) into LDS; capture h_{t-1} A-frags ----
    {
      const float4* xs4 = reinterpret_cast<const float4*>(xsrc + (size_t)t * D_IN);
      float4 f0 = xs4[0], f1 = xs4[1], f2 = xs4[2], f3 = xs4[3];
      if (t + 1 < T_STEPS)
        __builtin_prefetch(xsrc + (size_t)(t + 1) * D_IN, 0, 0);  // global_prefetch_b8
      union { _Float16 h[16]; uint4 q[2]; } u;
      u.h[0]=(_Float16)f0.x;  u.h[1]=(_Float16)f0.y;  u.h[2]=(_Float16)f0.z;  u.h[3]=(_Float16)f0.w;
      u.h[4]=(_Float16)f1.x;  u.h[5]=(_Float16)f1.y;  u.h[6]=(_Float16)f1.z;  u.h[7]=(_Float16)f1.w;
      u.h[8]=(_Float16)f2.x;  u.h[9]=(_Float16)f2.y;  u.h[10]=(_Float16)f2.z; u.h[11]=(_Float16)f2.w;
      u.h[12]=(_Float16)f3.x; u.h[13]=(_Float16)f3.y; u.h[14]=(_Float16)f3.z; u.h[15]=(_Float16)f3.w;
      reinterpret_cast<uint4*>(xdst)[0] = u.q[0];
      reinterpret_cast<uint4*>(xdst)[1] = u.q[1];
    }
    v16h ah1_0 = ldsA(h1buf, 64, 0, lane);
    v16h ah1_1 = ldsA(h1buf, 64, 1, lane);
    v16h ah2   = ldsA(h2buf, 32, 0, lane);
    v16h ah3   = ldsA(h3buf, 32, 0, lane);
    __syncthreads();

    // ---- phase 1: layer 1 (H=64), all waves; wave owns h1 cols [16*wv, 16*wv+16) ----
    {
      v8f a0 = {}, a1 = {}, a2 = {}, a3 = {};
#pragma unroll
      for (int kt = 0; kt < 4; ++kt) {                 // x @ Wih1^T, K=128
        v16h a = ldsA(xbuf, 128, kt, lane);
        a0 = wmma_f16(a, ldsB(Wf_ih1, kt * 16 +  0 + wv, lane), a0);
        a1 = wmma_f16(a, ldsB(Wf_ih1, kt * 16 +  4 + wv, lane), a1);
        a2 = wmma_f16(a, ldsB(Wf_ih1, kt * 16 +  8 + wv, lane), a2);
        a3 = wmma_f16(a, ldsB(Wf_ih1, kt * 16 + 12 + wv, lane), a3);
      }
#pragma unroll
      for (int kt = 0; kt < 2; ++kt) {                 // h1 @ Whh1^T, K=64
        v16h a = (kt == 0) ? ah1_0 : ah1_1;
        a0 = wmma_f16(a, bh1[kt][0], a0);
        a1 = wmma_f16(a, bh1[kt][1], a1);
        a2 = wmma_f16(a, bh1[kt][2], a2);
        a3 = wmma_f16(a, bh1[kt][3], a3);
      }
      const int col = (lane & 15) + 16 * wv;
      const float bi = bias1[col], bf = bias1[64 + col], bg = bias1[128 + col], bo = bias1[192 + col];
#pragma unroll
      for (int r = 0; r < 8; ++r) {
        float ig = fsigmoid(a0[r] + bi);
        float fg = fsigmoid(a1[r] + bf);
        float gg = ftanh   (a2[r] + bg);
        float og = fsigmoid(a3[r] + bo);
        float c  = fg * c1[r] + ig * gg;
        c1[r] = c;
        h1buf[(rbase + r) * 64 + col] = (_Float16)(og * ftanh(c));
      }
    }
    __syncthreads();

    // ---- phase 2: layer 2 (H=32), waves 0-1; wave owns h2 cols [16*wv, ...) ----
    if (wv < 2) {
      v8f a0 = {}, a1 = {}, a2 = {}, a3 = {};
#pragma unroll
      for (int kt = 0; kt < 2; ++kt) {                 // h1_t @ Wih2^T, K=64
        v16h a = ldsA(h1buf, 64, kt, lane);
        a0 = wmma_f16(a, ldsB(Wf_ih2, kt * 8 + 0 + wv, lane), a0);
        a1 = wmma_f16(a, ldsB(Wf_ih2, kt * 8 + 2 + wv, lane), a1);
        a2 = wmma_f16(a, ldsB(Wf_ih2, kt * 8 + 4 + wv, lane), a2);
        a3 = wmma_f16(a, ldsB(Wf_ih2, kt * 8 + 6 + wv, lane), a3);
      }
      // h2_{t-1} @ Whh2^T, K=32 (B-frags streamed from LDS)
      a0 = wmma_f16(ah2, ldsB(Wf_hh2, 0 + wv, lane), a0);
      a1 = wmma_f16(ah2, ldsB(Wf_hh2, 2 + wv, lane), a1);
      a2 = wmma_f16(ah2, ldsB(Wf_hh2, 4 + wv, lane), a2);
      a3 = wmma_f16(ah2, ldsB(Wf_hh2, 6 + wv, lane), a3);
      const int col = (lane & 15) + 16 * wv;
      const float bi = bias2[col], bf = bias2[32 + col], bg = bias2[64 + col], bo = bias2[96 + col];
#pragma unroll
      for (int r = 0; r < 8; ++r) {
        float ig = fsigmoid(a0[r] + bi);
        float fg = fsigmoid(a1[r] + bf);
        float gg = ftanh   (a2[r] + bg);
        float og = fsigmoid(a3[r] + bo);
        float c  = fg * c2[r] + ig * gg;
        c2[r] = c;
        h2buf[(rbase + r) * 32 + col] = (_Float16)(og * ftanh(c));
      }
    }
    __syncthreads();

    // ---- phase 3: layer 3 (H=16), wave 2 owns all 16 cols ----
    if (wv == 2) {
      v8f a0 = {}, a1 = {}, a2 = {}, a3 = {};
      v16h a = ldsA(h2buf, 32, 0, lane);               // h2_t @ Wih3^T, K=32
      a0 = wmma_f16(a, ldsB(Wf_ih3, 0, lane), a0);
      a1 = wmma_f16(a, ldsB(Wf_ih3, 1, lane), a1);
      a2 = wmma_f16(a, ldsB(Wf_ih3, 2, lane), a2);
      a3 = wmma_f16(a, ldsB(Wf_ih3, 3, lane), a3);
      // h3_{t-1} @ Whh3^T, K=16(+pad) (B-frags streamed from LDS)
      a0 = wmma_f16(ah3, ldsB(Wf_hh3, 0, lane), a0);
      a1 = wmma_f16(ah3, ldsB(Wf_hh3, 1, lane), a1);
      a2 = wmma_f16(ah3, ldsB(Wf_hh3, 2, lane), a2);
      a3 = wmma_f16(ah3, ldsB(Wf_hh3, 3, lane), a3);
      const int col = lane & 15;
      const float bi = bias3[col], bf = bias3[16 + col], bg = bias3[32 + col], bo = bias3[48 + col];
#pragma unroll
      for (int r = 0; r < 8; ++r) {
        float ig = fsigmoid(a0[r] + bi);
        float fg = fsigmoid(a1[r] + bf);
        float gg = ftanh   (a2[r] + bg);
        float og = fsigmoid(a3[r] + bo);
        float c  = fg * c3[r] + ig * gg;
        c3[r] = c;
        h3buf[(rbase + r) * 32 + col] = (_Float16)(og * ftanh(c));
      }
    }
    __syncthreads();
  }

  // ---- FC head on last timestep: out[b] = h3[b,:] . Wfc + bfc ----
  if (tid < B_TILE) {
    float s = bfc[0];
#pragma unroll
    for (int j = 0; j < 16; ++j)
      s = __builtin_fmaf((float)h3buf[tid * 32 + j], Wfc[j], s);
    out[b0 + tid] = s;
  }
}

extern "C" void kernel_launch(void* const* d_in, const int* in_sizes, int n_in,
                              void* d_out, int out_size, void* d_ws, size_t ws_size,
                              hipStream_t stream) {
  (void)n_in; (void)out_size; (void)d_ws; (void)ws_size;
  const float* x    = (const float*)d_in[0];
  const float* Wih1 = (const float*)d_in[1];
  const float* Whh1 = (const float*)d_in[2];
  const float* bih1 = (const float*)d_in[3];
  const float* bhh1 = (const float*)d_in[4];
  const float* Wih2 = (const float*)d_in[5];
  const float* Whh2 = (const float*)d_in[6];
  const float* bih2 = (const float*)d_in[7];
  const float* bhh2 = (const float*)d_in[8];
  const float* Wih3 = (const float*)d_in[9];
  const float* Whh3 = (const float*)d_in[10];
  const float* bih3 = (const float*)d_in[11];
  const float* bhh3 = (const float*)d_in[12];
  const float* Wfc  = (const float*)d_in[13];
  const float* bfc  = (const float*)d_in[14];
  float* out = (float*)d_out;

  const int nblocks = in_sizes[0] / (T_STEPS * D_IN * B_TILE);  // B / 16 = 64
  lstm3_fused_kernel<<<nblocks, TPB, 0, stream>>>(
      x, Wih1, Whh1, bih1, bhh1, Wih2, Whh2, bih2, bhh2,
      Wih3, Whh3, bih3, bhh3, Wfc, bfc, out);
}
